// MambaLayer_86354612453981
// MI455X (gfx1250) — matrix operand order repaired and checked by reading, not compiled, not run on hardware
//
#include <hip/hip_runtime.h>

#define BATCH 4
#define SEQ   32768
#define DIN   20
#define DI    20
#define NS    10
#define CH    128
#define NCH   (SEQ / CH)   // 256

typedef __attribute__((ext_vector_type(16))) __bf16        v16bf;
typedef __attribute__((ext_vector_type(8)))  float         v8f;
typedef __attribute__((ext_vector_type(8)))  unsigned int  v8u;
typedef __attribute__((ext_vector_type(4)))  float         f4;
typedef __attribute__((ext_vector_type(2)))  float         f2;

// ---- workspace layout (float offsets) ----
#define NU        (BATCH * SEQ * DI)            // 2621440
#define OFF_U     0
#define OFF_DELTA (NU)                          // 2621440
#define OFF_B     (2 * NU)                      // 5242880
#define OFF_C     (OFF_B + BATCH * SEQ * NS)    // 6553600
#define OFF_A     (OFF_C + BATCH * SEQ * NS)    // 7864320 (200)
#define OFF_BIASH (OFF_A + 256)                 // 7864576 (20)
#define OFF_FRAGS (OFF_BIASH + 256)             // 7864832 (2048 u32: 8 frags x 256)
#define OFF_APROD (OFF_FRAGS + 2048)            // 7866880
#define NAGG      (BATCH * NCH * DI * NS)       // 204800
#define OFF_SEND  (OFF_APROD + NAGG)
#define OFF_SINIT (OFF_SEND + NAGG)
#define WS_FLOATS (OFF_SINIT + NAGG)            // 8481280 floats ~ 32.4 MB

__device__ __forceinline__ unsigned int f2bf(float f) {
  union { float f; unsigned int u; } v; v.f = f;
  unsigned int u = v.u;
  unsigned int r = u + 0x7FFFu + ((u >> 16) & 1u);   // round-to-nearest-even
  return r >> 16;
}
__device__ __forceinline__ unsigned int packbf(float a, float b) {
  return f2bf(a) | (f2bf(b) << 16);
}

// =====================================================================
// Prep: Wcat = fold(w_in[:, :20], conv_k) ; bias_h ; A = -exp(A_log) ;
// weight fragments pre-swizzled to WMMA B layout (bf16).
// B-frag layout (16x16x32 bf16): dword[lane*8+j] = {K0=kb+(j<4 ? 2j : 16+2(j-4)), K0+1},
// n=lane%16, kb = (lane<16 ? 0 : 8).
// =====================================================================
__global__ __launch_bounds__(256) void mamba_prep(
    const float* __restrict__ w_in, const float* __restrict__ b_in,
    const float* __restrict__ conv_k, const float* __restrict__ conv_b,
    const float* __restrict__ w_x, const float* __restrict__ w_dt,
    const float* __restrict__ A_log,
    unsigned int* __restrict__ frags, float* __restrict__ bias_h,
    float* __restrict__ A_out)
{
  __shared__ float Wcat[64][20];
  const int tid = threadIdx.x;

  for (int idx = tid; idx < 64 * 20; idx += 256) {
    int k = idx / 20, o = idx % 20;
    float s = 0.f;
    if (k < 60) {
      int w = k / 20, c = k % 20;
      for (int i = 0; i < 20; ++i)
        s += w_in[c * 40 + i] * conv_k[w * 400 + i * 20 + o];
    }
    Wcat[k][o] = s;
  }
  for (int o = tid; o < 20; o += 256) {
    float s = conv_b[o];
    for (int w = 0; w < 3; ++w)
      for (int i = 0; i < 20; ++i)
        s += b_in[i] * conv_k[w * 400 + i * 20 + o];
    bias_h[o] = s;
  }
  for (int idx = tid; idx < DI * NS; idx += 256)
    A_out[idx] = -__expf(A_log[idx]);
  __syncthreads();

  for (int e = tid; e < 8 * 256; e += 256) {
    int f = e / 256, li = e % 256;
    int lane = li / 8, j = li % 8;
    int n  = lane & 15;
    int K0 = ((lane < 16) ? 0 : 8) + (j < 4 ? 2 * j : 16 + 2 * (j - 4));
    float e0 = 0.f, e1 = 0.f;
    if (f < 4) {                       // Wcat frags: f = ks*2 + nt
      int ks = f >> 1, nt = f & 1, N = n + nt * 16;
      int k0 = ks * 32 + K0;
      if (N < 20) { e0 = Wcat[k0][N]; e1 = Wcat[k0 + 1][N]; }
    } else if (f < 6) {                // w_x (20x25): f = 4 + nt
      int N = n + (f - 4) * 16;
      if (K0     < 20 && N < 25) e0 = w_x[K0 * 25 + N];
      if (K0 + 1 < 20 && N < 25) e1 = w_x[(K0 + 1) * 25 + N];
    } else {                           // w_dt (5x20): f = 6 + nt
      int N = n + (f - 6) * 16;
      if (K0     < 5 && N < 20) e0 = w_dt[K0 * 20 + N];
      if (K0 + 1 < 5 && N < 20) e1 = w_dt[(K0 + 1) * 20 + N];
    }
    frags[e] = packbf(e0, e1);
  }
}

// =====================================================================
// Frontend: one wave per 16-token tile. GEMM chain in WMMA bf16:
//   xcat(16x64)@Wcat(64x20) -> silu -> u ; u(16x32)@w_x(32x25) -> delta_raw|B|C
//   delta_raw(16x32)@w_dt(32x20) -> softmax -> delta
// A-frag (16x32 bf16): lane holds row=lane%16; dword j: K0 = klo + (j<4 ? 2j : 16+2(j-4)),
// klo = lane<16 ? 0 : 8.
// Fast path (t0 != 0): every 8-float K-run is 16B aligned -> global_load_b128.
// =====================================================================
__global__ __launch_bounds__(32) void mamba_frontend(
    const float* __restrict__ x, const float* __restrict__ b_x,
    const float* __restrict__ b_dt,
    const unsigned int* __restrict__ frags, const float* __restrict__ bias_h,
    float* __restrict__ u_out, float* __restrict__ delta_out,
    float* __restrict__ B_out, float* __restrict__ C_out)
{
  __shared__ float u_lds[16][20];
  __shared__ float xr_lds[16][8];
  __shared__ float dt_lds[16][20];

  const int lane = threadIdx.x;
  const int tile = blockIdx.x;
  const int b    = tile / (SEQ / 16);
  const int t0   = (tile % (SEQ / 16)) * 16;
  const int row  = lane & 15;
  const int hi   = lane >> 4;
  const int klo  = hi ? 8 : 0;
  const long batch0 = (long)b * SEQ * DIN;
  const long xbase  = ((long)b * SEQ + (t0 + row) - 2) * DIN;  // row starts at token-2

  __builtin_prefetch(x + batch0 + (long)t0 * DIN, 0, 0);       // global_prefetch_b8

  // ---- A frags for xcat (K = 0..63, only 0..59 valid, clipped at batch start)
  v8u apack[2] = {};
  if (t0 != 0) {
    // fast path: token-2 >= 14, all in-batch; vector loads (16B aligned).
    const float* base = x + xbase;
#pragma unroll
    for (int ks = 0; ks < 2; ++ks) {
      const float* r0 = base + ks * 32 + klo;
      f4 a0 = *(const f4*)(r0);
      f4 a1 = *(const f4*)(r0 + 4);
      apack[ks][0] = packbf(a0.x, a0.y);
      apack[ks][1] = packbf(a0.z, a0.w);
      apack[ks][2] = packbf(a1.x, a1.y);
      apack[ks][3] = packbf(a1.z, a1.w);
      const float* r1 = r0 + 16;
      f4 b0v = *(const f4*)(r1);
      apack[ks][4] = packbf(b0v.x, b0v.y);
      apack[ks][5] = packbf(b0v.z, b0v.w);
      if (ks == 0 || klo == 0) {       // skip K=60..63 (hi lanes, ks=1)
        f4 b1v = *(const f4*)(r1 + 4);
        apack[ks][6] = packbf(b1v.x, b1v.y);
        apack[ks][7] = packbf(b1v.z, b1v.w);
      }
    }
  } else {
    // slow path (first tile of each batch): per-element clip at batch start
#pragma unroll
    for (int ks = 0; ks < 2; ++ks) {
#pragma unroll
      for (int j = 0; j < 8; ++j) {
        int K0 = ks * 32 + klo + (j < 4 ? 2 * j : 16 + 2 * (j - 4));
        float e0 = 0.f, e1 = 0.f;
        long i0 = xbase + K0;
        if (K0     < 60 && i0     >= batch0) e0 = x[i0];
        if (K0 + 1 < 60 && i0 + 1 >= batch0) e1 = x[i0 + 1];
        apack[ks][j] = packbf(e0, e1);
      }
    }
  }

  // ---- conv GEMM: 2 ksteps x 2 ntiles
  v8f acc0 = {}, acc1 = {};
#pragma unroll
  for (int ks = 0; ks < 2; ++ks) {
    v16bf a  = __builtin_bit_cast(v16bf, apack[ks]);
    v8u  w0  = *(const v8u*)(frags + (ks * 2 + 0) * 256 + lane * 8);
    v8u  w1  = *(const v8u*)(frags + (ks * 2 + 1) * 256 + lane * 8);
    acc0 = __builtin_amdgcn_wmma_f32_16x16x32_bf16(
        false, a, false, __builtin_bit_cast(v16bf, w0), (short)0, acc0, false, false);
    acc1 = __builtin_amdgcn_wmma_f32_16x16x32_bf16(
        false, a, false, __builtin_bit_cast(v16bf, w1), (short)0, acc1, false, false);
  }

  // ---- bias + silu -> u (fast rcp sigmoid; store f32, stage in LDS)
  const int col0 = row, col1 = 16 + row;
  const float bh0 = bias_h[col0];
  const float bh1 = (col1 < DI) ? bias_h[col1] : 0.f;
#pragma unroll
  for (int j = 0; j < 8; ++j) {
    int r = j + hi * 8;
    long tok = (long)b * SEQ + t0 + r;
    float h0 = acc0[j] + bh0;
    float u0 = h0 * __builtin_amdgcn_rcpf(1.f + __expf(-h0));
    u_lds[r][col0] = u0;
    u_out[tok * DI + col0] = u0;
    if (col1 < DI) {
      float h1 = acc1[j] + bh1;
      float u1 = h1 * __builtin_amdgcn_rcpf(1.f + __expf(-h1));
      u_lds[r][col1] = u1;
      u_out[tok * DI + col1] = u1;
    }
  }
  __syncthreads();

  // ---- x-proj GEMM: u(16x32, K actual 20) @ w_x(32x25); f2 LDS loads
  v8u upack = {};
  {
    const f2* ur = (const f2*)&u_lds[row][0];
    const int fo = hi ? 4 : 0;          // K0 = klo + 2j  ->  f2 index klo/2 + j
#pragma unroll
    for (int j = 0; j < 4; ++j) { f2 t = ur[fo + j]; upack[j] = packbf(t.x, t.y); }
    if (!hi) {                          // lo lanes: K=16..19 valid (j=4,5)
      f2 t8 = ur[8]; f2 t9 = ur[9];
      upack[4] = packbf(t8.x, t8.y);
      upack[5] = packbf(t9.x, t9.y);
    }
  }
  v8f xacc0 = {}, xacc1 = {};
  {
    v16bf a = __builtin_bit_cast(v16bf, upack);
    v8u w0 = *(const v8u*)(frags + 4 * 256 + lane * 8);
    v8u w1 = *(const v8u*)(frags + 5 * 256 + lane * 8);
    xacc0 = __builtin_amdgcn_wmma_f32_16x16x32_bf16(
        false, a, false, __builtin_bit_cast(v16bf, w0), (short)0, xacc0, false, false);
    xacc1 = __builtin_amdgcn_wmma_f32_16x16x32_bf16(
        false, a, false, __builtin_bit_cast(v16bf, w1), (short)0, xacc1, false, false);
  }

  // ---- split xp -> delta_raw (LDS) | B | C (global)
  const float bx0 = b_x[col0];
  const float bx1 = (col1 < 25) ? b_x[col1] : 0.f;
#pragma unroll
  for (int j = 0; j < 8; ++j) {
    int r = j + hi * 8;
    long tok = (long)b * SEQ + t0 + r;
    float v0 = xacc0[j] + bx0;
    if (col0 < 5)        xr_lds[r][col0] = v0;
    else if (col0 < 15)  B_out[tok * NS + (col0 - 5)] = v0;
    else                 C_out[tok * NS + 0] = v0;
    if (col1 < 25)       C_out[tok * NS + (col1 - 15)] = xacc1[j] + bx1;
  }
  __syncthreads();

  // ---- dt GEMM: delta_raw(16x32, K actual 5) @ w_dt(32x20)
  v8u dpack = {};
  if (!hi) {
    f2 p0 = *(const f2*)&xr_lds[row][0];
    f2 p1 = *(const f2*)&xr_lds[row][2];
    dpack[0] = packbf(p0.x, p0.y);
    dpack[1] = packbf(p1.x, p1.y);
    dpack[2] = packbf(xr_lds[row][4], 0.f);
  }
  v8f dacc0 = {}, dacc1 = {};
  {
    v16bf a = __builtin_bit_cast(v16bf, dpack);
    v8u w0 = *(const v8u*)(frags + 6 * 256 + lane * 8);
    v8u w1 = *(const v8u*)(frags + 7 * 256 + lane * 8);
    dacc0 = __builtin_amdgcn_wmma_f32_16x16x32_bf16(
        false, a, false, __builtin_bit_cast(v16bf, w0), (short)0, dacc0, false, false);
    dacc1 = __builtin_amdgcn_wmma_f32_16x16x32_bf16(
        false, a, false, __builtin_bit_cast(v16bf, w1), (short)0, dacc1, false, false);
  }
  const float bd0 = b_dt[col0];
  const float bd1 = (col1 < DI) ? b_dt[col1] : 0.f;
#pragma unroll
  for (int j = 0; j < 8; ++j) {
    int r = j + hi * 8;
    dt_lds[r][col0] = dacc0[j] + bd0;
    if (col1 < DI) dt_lds[r][col1] = dacc1[j] + bd1;
  }
  __syncthreads();

  // ---- softmax over the 20 channels (one token per lane 0..15)
  if (lane < 16) {
    float m = -3.4e38f;
#pragma unroll
    for (int d = 0; d < DI; ++d) m = fmaxf(m, dt_lds[lane][d]);
    float ex[DI]; float sum = 0.f;
#pragma unroll
    for (int d = 0; d < DI; ++d) { ex[d] = __expf(dt_lds[lane][d] - m); sum += ex[d]; }
    float inv = __builtin_amdgcn_rcpf(sum);
    long tok = (long)b * SEQ + t0 + lane;
#pragma unroll
    for (int d = 0; d < DI; ++d) delta_out[tok * DI + d] = ex[d] * inv;
  }
}

// =====================================================================
// Pass 1: per-chunk local scan -> (aProd = exp(sum dA), sEnd). One thread per
// (b, chunk, d): owns 10 states, zero cross-lane traffic. f2 loads (8B rows).
// =====================================================================
__global__ __launch_bounds__(256) void scan_partial(
    const float* __restrict__ u, const float* __restrict__ delta,
    const float* __restrict__ Bm, const float* __restrict__ A,
    float* __restrict__ aProd, float* __restrict__ sEnd)
{
  int gid  = blockIdx.x * 256 + threadIdx.x;   // 0..20479 exactly
  int d    = gid % DI;
  int cidx = gid / DI;
  int ch   = cidx % NCH;
  int b    = cidx / NCH;

  float Ad[NS], s[NS], asum[NS];
#pragma unroll
  for (int n = 0; n < NS; ++n) { Ad[n] = A[d * NS + n]; s[n] = 0.f; asum[n] = 0.f; }

  int l0 = ch * CH;
  for (int l = l0; l < l0 + CH; ++l) {
    long base = (long)b * SEQ + l;
    float dlt = delta[base * DI + d];
    float du  = dlt * u[base * DI + d];
    const f2* Bp = (const f2*)(Bm + base * NS);
#pragma unroll
    for (int n2 = 0; n2 < NS / 2; ++n2) {
      f2 bb = Bp[n2];
      int n = 2 * n2;
      float dA0 = dlt * Ad[n],     dA1 = dlt * Ad[n + 1];
      asum[n] += dA0;              asum[n + 1] += dA1;
      s[n]     = __expf(dA0) * s[n]     + du * bb.x;
      s[n + 1] = __expf(dA1) * s[n + 1] + du * bb.y;
    }
  }
  long idx = ((long)(b * NCH + ch) * DI + d) * NS;
#pragma unroll
  for (int n = 0; n < NS; ++n) { aProd[idx + n] = __expf(asum[n]); sEnd[idx + n] = s[n]; }
}

// Pass 2: sequential combine over 256 chunk aggregates, one thread per (b,d,n).
__global__ __launch_bounds__(256) void scan_combine(
    const float* __restrict__ aProd, const float* __restrict__ sEnd,
    float* __restrict__ sInit)
{
  int b = blockIdx.x, t = threadIdx.x;
  if (t >= DI * NS) return;
  int d = t / NS, n = t % NS;
  float s = 0.f;
  for (int ch = 0; ch < NCH; ++ch) {
    long idx = ((long)(b * NCH + ch) * DI + d) * NS + n;
    sInit[idx] = s;
    s = aProd[idx] * s + sEnd[idx];
  }
}

// Pass 3: replay chunk from combined init state, emit y = sum_n s*C + u*D.
__global__ __launch_bounds__(256) void scan_final(
    const float* __restrict__ u, const float* __restrict__ delta,
    const float* __restrict__ Bm, const float* __restrict__ Cm,
    const float* __restrict__ A, const float* __restrict__ Dp,
    const float* __restrict__ sInit, float* __restrict__ out)
{
  int gid  = blockIdx.x * 256 + threadIdx.x;
  int d    = gid % DI;
  int cidx = gid / DI;
  int ch   = cidx % NCH;
  int b    = cidx / NCH;

  long sidx = ((long)(b * NCH + ch) * DI + d) * NS;
  float Ad[NS], s[NS];
#pragma unroll
  for (int n = 0; n < NS; ++n) { Ad[n] = A[d * NS + n]; s[n] = sInit[sidx + n]; }
  const float Dd = Dp[d];

  int l0 = ch * CH;
  for (int l = l0; l < l0 + CH; ++l) {
    long base = (long)b * SEQ + l;
    float dlt = delta[base * DI + d];
    float uu  = u[base * DI + d];
    float du  = dlt * uu;
    const f2* Bp = (const f2*)(Bm + base * NS);
    const f2* Cp = (const f2*)(Cm + base * NS);
    float y = 0.f;
#pragma unroll
    for (int n2 = 0; n2 < NS / 2; ++n2) {
      f2 bb = Bp[n2];
      f2 cc = Cp[n2];
      int n = 2 * n2;
      s[n]     = __expf(dlt * Ad[n])     * s[n]     + du * bb.x;
      s[n + 1] = __expf(dlt * Ad[n + 1]) * s[n + 1] + du * bb.y;
      y += s[n] * cc.x + s[n + 1] * cc.y;
    }
    out[base * DI + d] = y + uu * Dd;
  }
}

extern "C" void kernel_launch(void* const* d_in, const int* in_sizes, int n_in,
                              void* d_out, int out_size, void* d_ws, size_t ws_size,
                              hipStream_t stream) {
  (void)in_sizes; (void)n_in; (void)out_size;
  if (ws_size < (size_t)WS_FLOATS * sizeof(float)) return;

  const float* x       = (const float*)d_in[0];
  const float* w_in    = (const float*)d_in[1];
  const float* b_in    = (const float*)d_in[2];
  const float* conv_k  = (const float*)d_in[3];
  const float* conv_b  = (const float*)d_in[4];
  const float* w_x     = (const float*)d_in[5];
  const float* b_x     = (const float*)d_in[6];
  const float* w_dt    = (const float*)d_in[7];
  const float* b_dt    = (const float*)d_in[8];
  const float* A_log   = (const float*)d_in[9];
  const float* D_param = (const float*)d_in[10];

  float* ws = (float*)d_ws;
  float* wu     = ws + OFF_U;
  float* wdelta = ws + OFF_DELTA;
  float* wB     = ws + OFF_B;
  float* wC     = ws + OFF_C;
  float* wA     = ws + OFF_A;
  float* wbiash = ws + OFF_BIASH;
  unsigned int* wfrags = (unsigned int*)(ws + OFF_FRAGS);
  float* waProd = ws + OFF_APROD;
  float* wsEnd  = ws + OFF_SEND;
  float* wsInit = ws + OFF_SINIT;

  mamba_prep<<<1, 256, 0, stream>>>(w_in, b_in, conv_k, conv_b, w_x, w_dt, A_log,
                                    wfrags, wbiash, wA);
  mamba_frontend<<<BATCH * (SEQ / 16), 32, 0, stream>>>(
      x, b_x, b_dt, wfrags, wbiash, wu, wdelta, wB, wC);
  scan_partial<<<(BATCH * NCH * DI) / 256, 256, 0, stream>>>(wu, wdelta, wB, wA,
                                                             waProd, wsEnd);
  scan_combine<<<BATCH, 256, 0, stream>>>(waProd, wsEnd, wsInit);
  scan_final<<<(BATCH * NCH * DI) / 256, 256, 0, stream>>>(
      wu, wdelta, wB, wC, wA, D_param, wsInit, (float*)d_out);
}
